// MMRecBlock_44985487458345
// MI455X (gfx1250) — compile-verified, matrix operand order checked by reading, not run on hardware
//
#include <hip/hip_runtime.h>

// ============================================================================
// MMRecBlock for MI455X (gfx1250, wave32, WMMA bf16 + Tensor Data Mover)
//
// Pipeline:
//   rmsnorm -> q,k,v,z GEMMs -> gate MLPs (gelu/sigmoid epilogues) -> gamma
//   -> gz GEMM -> sequential scan (1 block/batch, L2-resident wgh)
//   -> mem K/V projections -> flash-attention (WMMA QK^T + PV, online softmax)
//   -> wao GEMM -> combine -> rmsnorm2 -> FFN (gelu) -> FFN2 (+residual) = out
//
// Matmuls: v_wmma_f32_16x16x32_bf16, fp32 accumulate.
// A-tiles (GEMM) and K-tiles (attention) are staged via TENSOR_LOAD_TO_LDS
// (D# descriptor with LDS padding to reproduce the padded tile layout),
// synchronized with s_wait_tensorcnt. Transposed tiles (B, V) stay on the
// manual ds_store path since the TDM does not transpose.
// ============================================================================

constexpr int kB  = 4;
constexpr int kS  = 2048;
constexpr int kD  = 1024;
constexpr int kH  = 16;
constexpr int kF  = 4096;
constexpr int kM  = 1024;
constexpr int kHD = 64;
constexpr int kRowsX = kB * kS;   // 8192
constexpr int kRowsM = kB * kM;   // 4096

typedef __bf16 bf16;
typedef __attribute__((ext_vector_type(16))) __bf16 v16bf;
typedef __attribute__((ext_vector_type(8)))  __bf16 v8bf;
typedef __attribute__((ext_vector_type(8)))  float  v8f;
typedef __attribute__((ext_vector_type(4)))  unsigned int u32x4;
typedef __attribute__((ext_vector_type(8)))  int i32x8;
typedef __attribute__((ext_vector_type(4)))  int i32x4;

union Frag { v16bf v; v8bf h[2]; };

__device__ __forceinline__ v8f vzero8() {
  v8f r;
#pragma unroll
  for (int i = 0; i < 8; ++i) r[i] = 0.f;
  return r;
}

__device__ __forceinline__ float gelu_tanh(float x) {
  const float c0 = 0.7978845608028654f;  // sqrt(2/pi)
  return 0.5f * x * (1.f + tanhf(c0 * (x + 0.044715f * x * x * x)));
}
__device__ __forceinline__ float sigmoidf(float x) {
  return 1.f / (1.f + __expf(-x));
}

// ---------------------------------------------------------------------------
// Tensor Data Mover: 2-D tile load (global -> LDS) with LDS row padding.
//   tile_k     : contiguous elements per row (bf16)
//   tile_rows  : number of rows
//   stride     : row stride in elements (tensor_dim0_stride)
//   pi_code    : pad_interval code (3 -> every 16 DWORDs, 4 -> every 32 DWORDs)
//   pa_code    : pad_amount code   (3 -> 4 DWORDs = 16 bytes)
// Issued by one wave; completion via s_wait_tensorcnt before the barrier.
// ---------------------------------------------------------------------------
__device__ __forceinline__ void tdm_raw(u32x4 g0, i32x8 g1, i32x4 g2, i32x4 g3) {
#if __has_include(<hip/amd_detail/amd_gfx1250_TDM.h>)
  i32x8 g4;
#pragma unroll
  for (int i = 0; i < 8; ++i) g4[i] = 0;
  __builtin_amdgcn_tensor_load_to_lds(g0, g1, g2, g3, g4, 0);
#else
  __builtin_amdgcn_tensor_load_to_lds(g0, g1, g2, g3, 0);
#endif
}

__device__ __forceinline__ void tdm_load_2d_bf16(unsigned lds_off,
                                                 const bf16* gptr,
                                                 unsigned tile_k,
                                                 unsigned tile_rows,
                                                 unsigned stride,
                                                 unsigned pi_code,
                                                 unsigned pa_code) {
  const unsigned long long ga = (unsigned long long)(size_t)gptr;
  u32x4 g0;
  g0[0] = 0x1u;                                    // count=1, user mode
  g0[1] = lds_off;                                 // lds_addr (bytes)
  g0[2] = (unsigned)(ga & 0xffffffffu);            // global_addr[31:0]
  g0[3] = (unsigned)((ga >> 32) & 0x01ffffffu)     // global_addr[56:32]
          | (2u << 30);                            // type = 2 ("image")
  i32x8 g1;
  g1[0] = (int)((1u << 16)                         // data_size = 2 bytes
                | (1u << 20)                       // pad_enable
                | (pi_code << 22) | (pa_code << 25));
  g1[1] = (int)((tile_k & 0xffffu) << 16);         // tensor_dim0[15:0]
  g1[2] = (int)(((tile_k >> 16) & 0xffffu)         // tensor_dim0[31:16]
                | ((tile_rows & 0xffffu) << 16));  // tensor_dim1[15:0]
  g1[3] = (int)(((tile_rows >> 16) & 0xffffu)      // tensor_dim1[31:16]
                | ((tile_k & 0xffffu) << 16));     // tile_dim0
  g1[4] = (int)(tile_rows & 0xffffu);              // tile_dim1 (tile_dim2=0)
  g1[5] = (int)stride;                             // tensor_dim0_stride[31:0]
  g1[6] = 0;
  g1[7] = 0;
  i32x4 g2, g3;
#pragma unroll
  for (int i = 0; i < 4; ++i) { g2[i] = 0; g3[i] = 0; }
  tdm_raw(g0, g1, g2, g3);
}

// ---------------------------------------------------------------------------
// Generic elementwise kernels
// ---------------------------------------------------------------------------
__global__ void cvt_f32_bf16(const float* __restrict__ s, bf16* __restrict__ d,
                             size_t n) {
  for (size_t i = (size_t)blockIdx.x * blockDim.x + threadIdx.x; i < n;
       i += (size_t)gridDim.x * blockDim.x)
    d[i] = (bf16)s[i];
}

__global__ void gamma_kernel(float* __restrict__ gb, const float* __restrict__ gm,
                             size_t n) {
  for (size_t i = (size_t)blockIdx.x * blockDim.x + threadIdx.x; i < n;
       i += (size_t)gridDim.x * blockDim.x)
    gb[i] = fminf(fmaxf(gb[i] * gm[i], 1e-6f), 1.f - 1e-6f);
}

__global__ void combine_kernel(const float* __restrict__ x,
                               const float* __restrict__ hseq,
                               const float* __restrict__ mo,
                               const float* __restrict__ v,
                               float* __restrict__ xres, size_t n) {
  for (size_t i = (size_t)blockIdx.x * blockDim.x + threadIdx.x; i < n;
       i += (size_t)gridDim.x * blockDim.x)
    xres[i] = x[i] + hseq[i] + mo[i] + 0.1f * v[i];
}

// ---------------------------------------------------------------------------
// RMSNorm (fp32 in -> bf16 out), one block per row of D=1024
// ---------------------------------------------------------------------------
__global__ __launch_bounds__(256) void rmsnorm_bf16(const float* __restrict__ x,
                                                    const float* __restrict__ g,
                                                    bf16* __restrict__ o) {
  const int r = blockIdx.x;
  const float* xr = x + (size_t)r * kD;
  float ss = 0.f;
  for (int i = threadIdx.x; i < kD; i += 256) {
    float v = xr[i];
    ss += v * v;
  }
  for (int off = 16; off > 0; off >>= 1) ss += __shfl_xor(ss, off, 32);
  __shared__ float red[8];
  const int lane = threadIdx.x & 31, wid = threadIdx.x >> 5;
  if (lane == 0) red[wid] = ss;
  __syncthreads();
  float tot = 0.f;
#pragma unroll
  for (int i = 0; i < 8; ++i) tot += red[i];
  const float inv = rsqrtf(tot / (float)kD + 1e-6f);
  bf16* orow = o + (size_t)r * kD;
  for (int i = threadIdx.x; i < kD; i += 256)
    orow[i] = (bf16)(xr[i] * inv * g[i]);
}

// ---------------------------------------------------------------------------
// WMMA bf16 GEMM:  C[rows x N] = act(A[rows x K] @ W[K x N] + bias) (+res)
// Block tile 128x64, 8 waves of 32x32, K-step 32.
// A tile staged by the Tensor Data Mover into padded [128][40] LDS layout;
// B tile staged transposed (N-major) by all threads.
// ---------------------------------------------------------------------------
template <typename OutT, int ACT, bool RES>
__global__ __launch_bounds__(256) void gemm_bf16(const bf16* __restrict__ A,
                                                 const bf16* __restrict__ W,
                                                 const float* __restrict__ bias,
                                                 const float* __restrict__ res,
                                                 OutT* __restrict__ C, int K,
                                                 int N) {
  const int tid = threadIdx.x;
  const int lane = tid & 31;
  const int wid = tid >> 5;
  const int wm = wid & 3;   // 0..3 -> 32-row group
  const int wn = wid >> 2;  // 0..1 -> 32-col group
  const int mBase = blockIdx.y * 128;
  const int nBase = blockIdx.x * 64;

  __shared__ bf16 lA[128][40];  // +8 pad, rows 80B (16B aligned)
  __shared__ bf16 lB[64][40];   // stored [n][k]

  v8f acc[2][2];
#pragma unroll
  for (int i = 0; i < 2; ++i)
#pragma unroll
    for (int j = 0; j < 2; ++j) acc[i][j] = vzero8();

  const int lr = lane & 15;
  const int half = lane >> 4;
  const int koff = half * 8;
  const unsigned ldsA = (unsigned)(size_t)&lA[0][0];

  for (int k0 = 0; k0 < K; k0 += 32) {
    __syncthreads();
    if (wid == 0) {  // TDM: A tile 128x32 bf16, pad 16B after each 64B row
      tdm_load_2d_bf16(ldsA, A + (size_t)mBase * K + k0,
                       /*tile_k=*/32, /*rows=*/128, /*stride=*/(unsigned)K,
                       /*pi=*/3, /*pa=*/3);
    }
    {  // stage B tile transposed: W[k0+kr][nBase+nb..+7] -> lB[n][k]
      const int kr = tid >> 3;
      const int nb = (tid & 7) * 8;
      const bf16* wp = W + (size_t)(k0 + kr) * N + nBase + nb;
      v8bf wv8 = *(const v8bf*)wp;
#pragma unroll
      for (int j = 0; j < 8; ++j) lB[nb + j][kr] = wv8[j];
      if (k0 + 32 < K) __builtin_prefetch(wp + (size_t)32 * N, 0, 1);
    }
    if (wid == 0) __builtin_amdgcn_s_wait_tensorcnt(0);
    __syncthreads();

#pragma unroll
    for (int i = 0; i < 2; ++i) {
      Frag a;
      const int ar = wm * 32 + i * 16 + lr;
      a.h[0] = *(const v8bf*)&lA[ar][koff];
      a.h[1] = *(const v8bf*)&lA[ar][koff + 16];
#pragma unroll
      for (int j = 0; j < 2; ++j) {
        Frag bfr;
        const int bn = wn * 32 + j * 16 + lr;
        bfr.h[0] = *(const v8bf*)&lB[bn][koff];
        bfr.h[1] = *(const v8bf*)&lB[bn][koff + 16];
        acc[i][j] = __builtin_amdgcn_wmma_f32_16x16x32_bf16(
            false, a.v, false, bfr.v, (short)0, acc[i][j], false, false);
      }
    }
  }

  // Epilogue: C fragment layout — lane: col = lane&15, rows = (lane>>4)*8 + e
#pragma unroll
  for (int i = 0; i < 2; ++i)
#pragma unroll
    for (int j = 0; j < 2; ++j) {
      const int gcol = nBase + wn * 32 + j * 16 + lr;
      const float bvl = bias[gcol];
#pragma unroll
      for (int e = 0; e < 8; ++e) {
        const int grow = mBase + wm * 32 + i * 16 + half * 8 + e;
        float val = acc[i][j][e] + bvl;
        if (ACT == 1) val = gelu_tanh(val);
        else if (ACT == 2) val = sigmoidf(val);
        if (RES) val += res[(size_t)grow * N + gcol];
        C[(size_t)grow * N + gcol] = (OutT)val;
      }
    }
}

// ---------------------------------------------------------------------------
// Flash attention with WMMA.
// Block = 8 waves; wave owns 16 query rows; block owns 128 rows of one (b,h).
// K tile staged by the TDM (padded [32][72]); V staged transposed manually.
// ---------------------------------------------------------------------------
__global__ __launch_bounds__(256) void attn_kernel(const bf16* __restrict__ q,
                                                   const bf16* __restrict__ kh,
                                                   const bf16* __restrict__ vh,
                                                   bf16* __restrict__ out) {
  const int tid = threadIdx.x;
  const int lane = tid & 31;
  const int wid = tid >> 5;
  const int lr = lane & 15;
  const int half = lane >> 4;
  const int koff = half * 8;

  const int qt = blockIdx.x;  // query tile (S/128)
  const int hh = blockIdx.y;  // head
  const int b = blockIdx.z;   // batch

  __shared__ bf16 lK[32][72];      // 32 keys x 64 hd (+pad)
  __shared__ bf16 lV[64][40];      // transposed: [hd][key]
  __shared__ bf16 lP[8][16][32];   // per-wave P staging

  // Q fragments (A layout: lane row = lr), two K-groups over hd=64
  const int qrow = qt * 128 + wid * 16 + lr;
  const bf16* qp = q + ((size_t)(b * kS + qrow)) * kD + hh * kHD;
  Frag qf[2];
#pragma unroll
  for (int kk = 0; kk < 2; ++kk) {
    qf[kk].h[0] = *(const v8bf*)(qp + kk * 32 + koff);
    qf[kk].h[1] = *(const v8bf*)(qp + kk * 32 + koff + 16);
  }

  v8f o4[4];
#pragma unroll
  for (int j = 0; j < 4; ++j) o4[j] = vzero8();
  float mrow[8], lrow[8];
#pragma unroll
  for (int e = 0; e < 8; ++e) { mrow[e] = -1e30f; lrow[e] = 0.f; }

  const unsigned ldsK = (unsigned)(size_t)&lK[0][0];

  for (int kc = 0; kc < kM; kc += 32) {
    __syncthreads();
    if (wid == 0) {  // TDM: K tile 32x64 bf16, pad 16B after each 128B row
      tdm_load_2d_bf16(ldsK,
                       kh + ((size_t)(b * kM + kc)) * kD + hh * kHD,
                       /*tile_k=*/64, /*rows=*/32, /*stride=*/(unsigned)kD,
                       /*pi=*/4, /*pa=*/3);
    }
    {  // stage V (transposed) — TDM cannot transpose
      const int key = tid >> 3;
      const int cb = (tid & 7) * 8;
      const size_t rbase = ((size_t)(b * kM + kc + key)) * kD + hh * kHD + cb;
      v8bf vv = *(const v8bf*)(vh + rbase);
#pragma unroll
      for (int j = 0; j < 8; ++j) lV[cb + j][key] = vv[j];
    }
    if (wid == 0) __builtin_amdgcn_s_wait_tensorcnt(0);
    __syncthreads();

    // scores: two 16x16 tiles over the 32-key chunk
    v8f sc[2];
    sc[0] = vzero8();
    sc[1] = vzero8();
#pragma unroll
    for (int j = 0; j < 2; ++j)
#pragma unroll
      for (int kk = 0; kk < 2; ++kk) {
        Frag bfr;
        bfr.h[0] = *(const v8bf*)&lK[j * 16 + lr][kk * 32 + koff];
        bfr.h[1] = *(const v8bf*)&lK[j * 16 + lr][kk * 32 + koff + 16];
        sc[j] = __builtin_amdgcn_wmma_f32_16x16x32_bf16(
            false, qf[kk].v, false, bfr.v, (short)0, sc[j], false, false);
      }

    // online softmax (rows live on 16-lane halves; xor masks < 16 stay inside)
#pragma unroll
    for (int e = 0; e < 8; ++e) {
      float s0 = sc[0][e] * 0.125f;  // 1/sqrt(64)
      float s1 = sc[1][e] * 0.125f;
      float rm = fmaxf(s0, s1);
      for (int off = 1; off < 16; off <<= 1) rm = fmaxf(rm, __shfl_xor(rm, off, 32));
      const float mn = fmaxf(mrow[e], rm);
      const float alpha = __expf(mrow[e] - mn);
      const float p0 = __expf(s0 - mn);
      const float p1 = __expf(s1 - mn);
      float rs = p0 + p1;
      for (int off = 1; off < 16; off <<= 1) rs += __shfl_xor(rs, off, 32);
      lrow[e] = lrow[e] * alpha + rs;
      mrow[e] = mn;
#pragma unroll
      for (int j = 0; j < 4; ++j) o4[j][e] *= alpha;
      const int r = half * 8 + e;
      lP[wid][r][lr] = (bf16)p0;
      lP[wid][r][16 + lr] = (bf16)p1;
    }
    asm volatile("s_wait_dscnt 0" ::: "memory");  // P visible wave-wide

    // O += P @ Vc
    Frag pf;
    pf.h[0] = *(const v8bf*)&lP[wid][lr][koff];
    pf.h[1] = *(const v8bf*)&lP[wid][lr][koff + 16];
#pragma unroll
    for (int j = 0; j < 4; ++j) {
      Frag vf;
      vf.h[0] = *(const v8bf*)&lV[j * 16 + lr][koff];
      vf.h[1] = *(const v8bf*)&lV[j * 16 + lr][koff + 16];
      o4[j] = __builtin_amdgcn_wmma_f32_16x16x32_bf16(
          false, pf.v, false, vf.v, (short)0, o4[j], false, false);
    }
  }

#pragma unroll
  for (int j = 0; j < 4; ++j)
#pragma unroll
    for (int e = 0; e < 8; ++e) {
      const int r = half * 8 + e;
      const int srow = qt * 128 + wid * 16 + r;
      const int c = j * 16 + lr;
      out[((size_t)(b * kS + srow)) * kD + hh * kHD + c] =
          (bf16)(o4[j][e] / lrow[e]);
    }
}

// ---------------------------------------------------------------------------
// Sequential recurrence. One block per batch, 1024 threads (one per channel).
// wgh (4 MB fp32) is L2-resident; reads of wgh[k*D + d] are coalesced in d.
// ---------------------------------------------------------------------------
__global__ __launch_bounds__(1024) void scan_kernel(
    const float* __restrict__ gz, const bf16* __restrict__ z,
    const float* __restrict__ gamma, const float* __restrict__ wgh,
    const float* __restrict__ bgh, float* __restrict__ hseq) {
  __shared__ float h[kD];
  const int b = blockIdx.x;
  const int d = threadIdx.x;
  h[d] = 0.f;
  __syncthreads();
  const float bghd = bgh[d];
  for (int t = 0; t < kS; ++t) {
    const size_t row = (size_t)(b * kS + t) * kD;
    float acc = gz[row + d] + bghd;
#pragma unroll 4
    for (int k = 0; k < kD; ++k) acc = fmaf(h[k], wgh[(size_t)k * kD + d], acc);
    const float gate = sigmoidf(acc);
    const float hp = h[d];
    float ht = (1.f - gate) * hp + gate * (float)z[row + d] + gamma[row + d] * hp;
    ht = fminf(fmaxf(ht, -100.f), 100.f);
    __syncthreads();
    h[d] = ht;
    hseq[row + d] = ht;
    __syncthreads();
  }
}

// ---------------------------------------------------------------------------
// Host orchestration
// ---------------------------------------------------------------------------
extern "C" void kernel_launch(void* const* d_in, const int* in_sizes, int n_in,
                              void* d_out, int out_size, void* d_ws,
                              size_t ws_size, hipStream_t stream) {
  (void)in_sizes; (void)n_in; (void)out_size; (void)ws_size;
  const float* x    = (const float*)d_in[0];
  const float* memk = (const float*)d_in[1];
  const float* memv = (const float*)d_in[2];
  const float* n1   = (const float*)d_in[3];
  const float* n2   = (const float*)d_in[4];
  const float* wq  = (const float*)d_in[5];  const float* bq  = (const float*)d_in[6];
  const float* wk  = (const float*)d_in[7];  const float* bk  = (const float*)d_in[8];
  const float* wv  = (const float*)d_in[9];  const float* bv  = (const float*)d_in[10];
  const float* wz  = (const float*)d_in[11]; const float* bz  = (const float*)d_in[12];
  const float* wgz = (const float*)d_in[13]; const float* bgz = (const float*)d_in[14];
  const float* wgh = (const float*)d_in[15]; const float* bgh = (const float*)d_in[16];
  const float* wg1 = (const float*)d_in[17]; const float* bg1 = (const float*)d_in[18];
  const float* wg2 = (const float*)d_in[19]; const float* bg2 = (const float*)d_in[20];
  const float* wc1 = (const float*)d_in[21]; const float* bc1 = (const float*)d_in[22];
  const float* wc2 = (const float*)d_in[23]; const float* bc2 = (const float*)d_in[24];
  const float* wak = (const float*)d_in[25]; const float* bak = (const float*)d_in[26];
  const float* wav = (const float*)d_in[27]; const float* bav = (const float*)d_in[28];
  const float* wao = (const float*)d_in[29]; const float* bao = (const float*)d_in[30];
  const float* w1  = (const float*)d_in[31]; const float* b1  = (const float*)d_in[32];
  const float* w2  = (const float*)d_in[33]; const float* b2  = (const float*)d_in[34];
  float* out = (float*)d_out;

  char* base = (char*)d_ws;
  size_t off = 0;
  auto alloc = [&](size_t bytes) -> void* {
    void* p = base + off;
    off = (off + bytes + 255) & ~(size_t)255;
    return p;
  };

  const size_t nBSD  = (size_t)kRowsX * kD;
  const size_t nBMD  = (size_t)kRowsM * kD;
  const size_t nBSD4 = (size_t)kRowsX * (kD / 4);
  const size_t nBSF  = (size_t)kRowsX * kF;

  bf16*  xnorm  = (bf16*)alloc(nBSD * 2);
  bf16*  qb     = (bf16*)alloc(nBSD * 2);
  bf16*  kb     = (bf16*)alloc(nBSD * 2);
  bf16*  zb     = (bf16*)alloc(nBSD * 2);
  float* vf     = (float*)alloc(nBSD * 4);
  bf16*  g1b    = (bf16*)alloc(nBSD4 * 2);
  bf16*  c1b    = (bf16*)alloc(nBSD4 * 2);
  float* gbase  = (float*)alloc(nBSD * 4);  // becomes gamma in-place
  float* gmod   = (float*)alloc(nBSD * 4);
  float* gzf    = (float*)alloc(nBSD * 4);
  float* hseq   = (float*)alloc(nBSD * 4);
  bf16*  mkb    = (bf16*)alloc(nBMD * 2);
  bf16*  mvb    = (bf16*)alloc(nBMD * 2);
  bf16*  khb    = (bf16*)alloc(nBMD * 2);
  bf16*  vhb    = (bf16*)alloc(nBMD * 2);
  bf16*  attnb  = (bf16*)alloc(nBSD * 2);
  float* mout   = (float*)alloc(nBSD * 4);
  float* xres   = (float*)alloc(nBSD * 4);
  bf16*  xnorm2 = (bf16*)alloc(nBSD * 2);
  bf16*  ffn1   = (bf16*)alloc(nBSF * 2);
  // bf16 weight shadows
  const size_t nDD = (size_t)kD * kD, nDQ = (size_t)kD * (kD / 4),
               nDF = (size_t)kD * kF;
  bf16* wq_b  = (bf16*)alloc(nDD * 2);
  bf16* wk_b  = (bf16*)alloc(nDD * 2);
  bf16* wv_b  = (bf16*)alloc(nDD * 2);
  bf16* wz_b  = (bf16*)alloc(nDD * 2);
  bf16* wgz_b = (bf16*)alloc(nDD * 2);
  bf16* wak_b = (bf16*)alloc(nDD * 2);
  bf16* wav_b = (bf16*)alloc(nDD * 2);
  bf16* wao_b = (bf16*)alloc(nDD * 2);
  bf16* wg1_b = (bf16*)alloc(nDQ * 2);
  bf16* wg2_b = (bf16*)alloc(nDQ * 2);
  bf16* wc1_b = (bf16*)alloc(nDQ * 2);
  bf16* wc2_b = (bf16*)alloc(nDQ * 2);
  bf16* w1_b  = (bf16*)alloc(nDF * 2);
  bf16* w2_b  = (bf16*)alloc(nDF * 2);

  auto cvt = [&](const float* s, bf16* d, size_t n) {
    int blocks = (int)((n + 2047) / 2048);
    if (blocks > 8192) blocks = 8192;
    cvt_f32_bf16<<<blocks, 256, 0, stream>>>(s, d, n);
  };
  cvt(wq, wq_b, nDD);   cvt(wk, wk_b, nDD);   cvt(wv, wv_b, nDD);
  cvt(wz, wz_b, nDD);   cvt(wgz, wgz_b, nDD); cvt(wak, wak_b, nDD);
  cvt(wav, wav_b, nDD); cvt(wao, wao_b, nDD);
  cvt(wg1, wg1_b, nDQ); cvt(wg2, wg2_b, nDQ);
  cvt(wc1, wc1_b, nDQ); cvt(wc2, wc2_b, nDQ);
  cvt(w1, w1_b, nDF);   cvt(w2, w2_b, nDF);
  cvt(memk, mkb, nBMD); cvt(memv, mvb, nBMD);

  const dim3 blk(256);
  rmsnorm_bf16<<<kRowsX, blk, 0, stream>>>(x, n1, xnorm);

  const dim3 gD(kD / 64, kRowsX / 128);        // [8192 x 1024] outputs
  const dim3 gQ4((kD / 4) / 64, kRowsX / 128); // [8192 x 256]
  const dim3 gMm(kD / 64, kRowsM / 128);       // [4096 x 1024]
  const dim3 gF1(kF / 64, kRowsX / 128);       // [8192 x 4096]

  gemm_bf16<bf16, 0, false><<<gD, blk, 0, stream>>>(xnorm, wq_b, bq, nullptr, qb, kD, kD);
  gemm_bf16<bf16, 0, false><<<gD, blk, 0, stream>>>(xnorm, wk_b, bk, nullptr, kb, kD, kD);
  gemm_bf16<float, 0, false><<<gD, blk, 0, stream>>>(xnorm, wv_b, bv, nullptr, vf, kD, kD);
  gemm_bf16<bf16, 0, false><<<gD, blk, 0, stream>>>(xnorm, wz_b, bz, nullptr, zb, kD, kD);

  gemm_bf16<bf16, 1, false><<<gQ4, blk, 0, stream>>>(zb, wg1_b, bg1, nullptr, g1b, kD, kD / 4);
  gemm_bf16<float, 2, false><<<gD, blk, 0, stream>>>(g1b, wg2_b, bg2, nullptr, gbase, kD / 4, kD);
  gemm_bf16<bf16, 1, false><<<gQ4, blk, 0, stream>>>(kb, wc1_b, bc1, nullptr, c1b, kD, kD / 4);
  gemm_bf16<float, 2, false><<<gD, blk, 0, stream>>>(c1b, wc2_b, bc2, nullptr, gmod, kD / 4, kD);
  gamma_kernel<<<4096, blk, 0, stream>>>(gbase, gmod, nBSD);

  gemm_bf16<float, 0, false><<<gD, blk, 0, stream>>>(zb, wgz_b, bgz, nullptr, gzf, kD, kD);
  scan_kernel<<<kB, 1024, 0, stream>>>(gzf, zb, gbase, wgh, bgh, hseq);

  gemm_bf16<bf16, 0, false><<<gMm, blk, 0, stream>>>(mkb, wak_b, bak, nullptr, khb, kD, kD);
  gemm_bf16<bf16, 0, false><<<gMm, blk, 0, stream>>>(mvb, wav_b, bav, nullptr, vhb, kD, kD);

  const dim3 gA(kS / 128, kH, kB);
  attn_kernel<<<gA, blk, 0, stream>>>(qb, khb, vhb, attnb);
  gemm_bf16<float, 0, false><<<gD, blk, 0, stream>>>(attnb, wao_b, bao, nullptr, mout, kD, kD);

  combine_kernel<<<4096, blk, 0, stream>>>(x, hseq, mout, vf, xres, nBSD);
  rmsnorm_bf16<<<kRowsX, blk, 0, stream>>>(xres, n2, xnorm2);

  gemm_bf16<bf16, 1, false><<<gF1, blk, 0, stream>>>(xnorm2, w1_b, b1, nullptr, ffn1, kD, kF);
  gemm_bf16<float, 0, true><<<gD, blk, 0, stream>>>(ffn1, w2_b, b2, xres, out, kF, kD);
}
